// AGFL_8924942042041
// MI455X (gfx1250) — matrix operand order repaired and checked by reading, not compiled, not run on hardware
//
#include <hip/hip_runtime.h>

// ---------------- types ----------------
typedef __attribute__((ext_vector_type(16))) __bf16 v16bf;
typedef __attribute__((ext_vector_type(8)))  __bf16 v8bf;
typedef __attribute__((ext_vector_type(8)))  float  v8f;
typedef __attribute__((ext_vector_type(4)))  float  v4f;
typedef __attribute__((ext_vector_type(4)))  unsigned int v4u;
typedef __attribute__((ext_vector_type(2)))  unsigned int v2u;
typedef __attribute__((ext_vector_type(4)))  unsigned int u32x4;
typedef __attribute__((ext_vector_type(8)))  int i32x8;
typedef __attribute__((ext_vector_type(4)))  int i32x4;

#define B_   4
#define N_   1024
#define D_   512
#define H_   8
#define DH_  64
#define BH_  (B_ * H_)
#define KC_  128          // hop-GEMM K panel staged in LDS
#define LDB_ 136          // LDS row stride in ushorts (64 dw data + 4 dw TDM pad)

// ---------------- TDM availability ----------------
#if defined(__has_include)
#if __has_include(<hip/amd_detail/amd_gfx1250_TDM.h>)
#define TDM_ARITY6 1
#endif
#endif
#if defined(__has_builtin)
#if __has_builtin(__builtin_amdgcn_tensor_load_to_lds) && \
    __has_builtin(__builtin_amdgcn_s_wait_tensorcnt)
#define HAVE_TDM 1
#endif
#endif
#ifndef HAVE_TDM
#define HAVE_TDM 0
#endif

// ---------------- helpers ----------------
__device__ __forceinline__ unsigned short f2bf(float f) {
  unsigned u = __float_as_uint(f);
  u = u + 0x7fffu + ((u >> 16) & 1u);   // round-to-nearest-even
  return (unsigned short)(u >> 16);
}

// A-fragment (16x32 bf16): per lane two contiguous 8-elem chunks at +0 and +8
__device__ __forceinline__ v16bf joinA(const unsigned short* p) {
  v8bf lo = *(const v8bf*)(const void*)p;
  v8bf hi = *(const v8bf*)(const void*)(p + 8);
  v16bf r;
#pragma unroll
  for (int i = 0; i < 8; ++i) { r[i] = lo[i]; r[i + 8] = hi[i]; }
  return r;
}

// B-fragment (32x16 bf16): one contiguous 16-elem chunk, 32B-aligned source
__device__ __forceinline__ v16bf loadB(const unsigned short* p) {
  return *(const v16bf*)(const void*)p;
}

// B-fragment from 16B-aligned source (LDS panel rows are 272B-strided)
__device__ __forceinline__ v16bf loadB16(const unsigned short* p) {
  v8bf lo = *(const v8bf*)(const void*)p;
  v8bf hi = *(const v8bf*)(const void*)(p + 8);
  v16bf r;
#pragma unroll
  for (int i = 0; i < 8; ++i) { r[i] = lo[i]; r[i + 8] = hi[i]; }
  return r;
}

__device__ __forceinline__ v8f wmma_bf16(v16bf a, v16bf b, v8f c) {
  return __builtin_amdgcn_wmma_f32_16x16x32_bf16(false, a, false, b,
                                                 (short)0, c, false, false);
}

#if HAVE_TDM
// Issue a 2-D TDM load: tile_d0 x tile_d1 elements (2 bytes each) from a
// row-major tensor (dim0 contiguous, stride_d0 elements between rows) into
// LDS with 4-DWORD padding appended after every 64 DWORDs (one 128-elem row).
// D# encoding per CDNA5 ISA 8.3/8.4.
__device__ __forceinline__ void tdm_load_2d(const void* gsrc, unsigned lds_byte,
                                            unsigned tile_d0, unsigned tile_d1,
                                            unsigned tensor_d0, unsigned tensor_d1,
                                            unsigned stride_d0) {
  unsigned long long ga = (unsigned long long)(__SIZE_TYPE__)gsrc;
  u32x4 g0;
  g0[0] = 1u;                                         // count=1, user mode
  g0[1] = lds_byte;                                   // lds_addr (bytes)
  g0[2] = (unsigned)(ga & 0xffffffffu);               // global_addr[31:0]
  g0[3] = ((unsigned)(ga >> 32) & 0x01ffffffu)        // global_addr[56:32]
          | 0x80000000u;                              // type=2 ("image")
  i32x8 g1;
  g1[0] = (int)((1u << 16)        // data_size = 2 bytes
              | (1u << 20)        // pad_enable
              | (5u << 22)        // pad_interval: 2^(5+1)=64 DWORDs
              | (3u << 25));      // pad_amount: 4 DWORDs
  g1[1] = (int)((tensor_d0 & 0xffffu) << 16);         // tensor_dim0 lo16
  g1[2] = (int)((tensor_d0 >> 16) | ((tensor_d1 & 0xffffu) << 16));
  g1[3] = (int)((tensor_d1 >> 16) | (tile_d0 << 16)); // tile_dim0
  g1[4] = (int)(tile_d1 & 0xffffu);                   // tile_dim1 (tile_dim2=0)
  g1[5] = (int)stride_d0;                             // tensor_dim0_stride lo32
  g1[6] = 0;
  g1[7] = 0;
  i32x4 z4 = {0, 0, 0, 0};
#ifdef TDM_ARITY6
  i32x8 z8 = {0, 0, 0, 0, 0, 0, 0, 0};
  __builtin_amdgcn_tensor_load_to_lds(g0, g1, z4, z4, z8, 0);
#else
  __builtin_amdgcn_tensor_load_to_lds(g0, g1, z4, z4, 0);
#endif
}
#endif

// ---------------- pack / convert kernels ----------------
__global__ void pack_x(const float* __restrict__ X,
                       unsigned short* __restrict__ Xh,
                       unsigned short* __restrict__ XhT) {
  int i = blockIdx.x * 256 + threadIdx.x;          // < B*N*D
  int c = i % D_;
  int n = (i / D_) % N_;
  int b = i / (D_ * N_);
  int h = c / DH_, dh = c % DH_;
  unsigned short v = f2bf(X[i]);
  size_t bh = (size_t)(b * H_ + h);
  Xh[(bh * N_ + n) * DH_ + dh] = v;                // [bh][n][dh] row-major
  XhT[bh * DH_ * N_ + (size_t)dh * N_ + n] = v;    // [bh][dh][n] transposed
}

__global__ void pack_wf(const float* __restrict__ Wf,
                        const float* __restrict__ alpha,
                        unsigned short* __restrict__ Wts) {
  int i = blockIdx.x * 256 + threadIdx.x;          // < H*4*DH*DH, Wf[h][k][d][e]
  int e = i % DH_;
  int d = (i / DH_) % DH_;
  int hk = i / (DH_ * DH_);
  float v = alpha[hk] * Wf[i];                     // fold alpha into weights
  Wts[(size_t)hk * DH_ * DH_ + (size_t)e * DH_ + d] = f2bf(v);  // store Wt[e][d]
}

__global__ void pack_wp(const float* __restrict__ Wp_in,
                        unsigned short* __restrict__ Wp) {
  int i = blockIdx.x * 256 + threadIdx.x;          // < D*D, W_proj[o][c]
  Wp[i] = f2bf(Wp_in[i]);
}

__global__ void cvt_h(const float* __restrict__ Ha, unsigned short* __restrict__ Hb) {
  int i = blockIdx.x * 256 + threadIdx.x;          // < B*N*D
  Hb[i] = f2bf(Ha[i]);
}

// ---------------- S = (Xh Xh^T) / (8 * tau) ----------------
// grid (128, BH), block 256 (8 waves). Each wave -> 16x64 output tile.
__global__ void score_gemm(const unsigned short* __restrict__ Xh,
                           const float* __restrict__ temperature,
                           float* __restrict__ S) {
  const int lane = threadIdx.x & 31, wv = threadIdx.x >> 5;
  const int l16 = lane & 15, hp = lane >> 4;
  const int bh = blockIdx.y;
  const int tl = blockIdx.x * 8 + wv;              // [0,1024)
  const int m0 = (tl >> 4) * 16;
  const int n0 = (tl & 15) * 64;
  const unsigned short* Xbh = Xh + (size_t)bh * N_ * DH_;

  v8f zero = {};
  v8f acc[4] = {zero, zero, zero, zero};
#pragma unroll
  for (int k0 = 0; k0 < DH_; k0 += 32) {
    v16bf a = joinA(Xbh + (size_t)(m0 + l16) * DH_ + k0 + hp * 16);
#pragma unroll
    for (int j = 0; j < 4; ++j) {
      v16bf b = loadB(Xbh + (size_t)(n0 + j * 16 + l16) * DH_ + k0 + hp * 16);
      acc[j] = wmma_bf16(a, b, acc[j]);
    }
  }
  float tau = temperature[bh % H_];
  tau = fminf(fmaxf(tau, 0.1f), 5.0f);
  float sc = __builtin_amdgcn_rcpf(8.0f * tau);    // 8 = sqrt(DH)
  float* Sr = S + (size_t)bh * N_ * N_;
#pragma unroll
  for (int j = 0; j < 4; ++j)
#pragma unroll
    for (int r = 0; r < 8; ++r)
      Sr[(size_t)(m0 + r + hp * 8) * N_ + (n0 + j * 16 + l16)] = acc[j][r] * sc;
}

// ---------------- exact top-k threshold + softmax, A written bf16 in place ---
// one block (256 thr) per row; radix-select k-th largest via monotone uint map.
__global__ void softmax_topk(float* __restrict__ S,
                             const int* __restrict__ pli,
                             const int* __restrict__ pL) {
  __shared__ int   ired[256];
  __shared__ float fred[256];
  const int tid = threadIdx.x;
  float* Sr = S + (size_t)blockIdx.x * N_;

  v4f v = ((const v4f*)Sr)[tid];
  float vals[4] = {v[0], v[1], v[2], v[3]};
  unsigned um[4];
#pragma unroll
  for (int i = 0; i < 4; ++i) {
    unsigned u = __float_as_uint(vals[i]);
    um[i] = (u & 0x80000000u) ? ~u : (u | 0x80000000u);
  }

  int li = pli[0], L = pL[0];
  float sp = 0.8f + (0.2f - 0.8f) * __expf(-3.0f * (float)li / (float)L);
  int kv = (int)((1.0f - sp) * (float)N_);
  if (kv < 1) kv = 1;
  const bool doThresh = (kv < N_);

  unsigned T = 0u;
  if (doThresh) {
    for (int bit = 31; bit >= 0; --bit) {
      unsigned cand = T | (1u << bit);
      int c = 0;
#pragma unroll
      for (int i = 0; i < 4; ++i) c += (um[i] >= cand) ? 1 : 0;
      ired[tid] = c; __syncthreads();
      for (int s = 128; s > 0; s >>= 1) {
        if (tid < s) ired[tid] += ired[tid + s];
        __syncthreads();
      }
      if (ired[0] >= kv) T = cand;
      __syncthreads();
    }
  }
  float thresh = doThresh
      ? __uint_as_float((T & 0x80000000u) ? (T ^ 0x80000000u) : ~T)
      : -__builtin_inff();

  // row max
  float mx = fmaxf(fmaxf(vals[0], vals[1]), fmaxf(vals[2], vals[3]));
  fred[tid] = mx; __syncthreads();
  for (int s = 128; s > 0; s >>= 1) {
    if (tid < s) fred[tid] = fmaxf(fred[tid], fred[tid + s]);
    __syncthreads();
  }
  mx = fred[0]; __syncthreads();

  float p[4]; float lsum = 0.0f;
#pragma unroll
  for (int i = 0; i < 4; ++i) {
    p[i] = (vals[i] >= thresh) ? __expf(vals[i] - mx) : 0.0f;
    lsum += p[i];
  }
  fred[tid] = lsum; __syncthreads();
  for (int s = 128; s > 0; s >>= 1) {
    if (tid < s) fred[tid] += fred[tid + s];
    __syncthreads();
  }
  float inv = 1.0f / fred[0];

  // write bf16 A row aliased over the first half of this fp32 row (safe: this
  // block owns the row and has the full row in registers). ld becomes 2048.
  unsigned short* Ar = (unsigned short*)Sr;
  v2u pk;
  pk[0] = (unsigned)f2bf(p[0] * inv) | ((unsigned)f2bf(p[1] * inv) << 16);
  pk[1] = (unsigned)f2bf(p[2] * inv) | ((unsigned)f2bf(p[3] * inv) << 16);
  ((v2u*)Ar)[tid] = pk;
}

// ---------------- Pnext = A @ P  (1024 x 64 x 1024, bf16) ----------------
// A is bf16 aliased in S (row stride 2048 elems). The shared B panel
// (PT[0:64, kp:kp+128], 16KB) is staged into LDS once per workgroup via the
// Tensor Data Mover (fallback: cooperative copy), with 4-DWORD row padding so
// the 16 ds-read lanes hit distinct banks. Output written in both layouts.
__global__ void hop_gemm(const float* __restrict__ Sbase,
                         const unsigned short* __restrict__ PTin,
                         unsigned short* __restrict__ PrOut,
                         unsigned short* __restrict__ PTout) {
  __shared__ unsigned short Bsm[DH_ * LDB_];       // 64 rows x 136 (17KB)
  const int lane = threadIdx.x & 31, wv = threadIdx.x >> 5;
  const int l16 = lane & 15, hp = lane >> 4;
  const int bh = blockIdx.y;
  const int m0 = (blockIdx.x * 8 + wv) * 16;
  const unsigned short* Abh =
      (const unsigned short*)(const void*)(Sbase + (size_t)bh * N_ * N_);
  const unsigned short* PTb = PTin + (size_t)bh * DH_ * N_;

  v8f zero = {};
  v8f acc[4] = {zero, zero, zero, zero};

  for (int kp = 0; kp < N_; kp += KC_) {
#if HAVE_TDM
    if (wv == 0) {
      tdm_load_2d(PTb + kp, (unsigned)(__SIZE_TYPE__)&Bsm[0],
                  KC_, DH_, N_, DH_, N_);
      __builtin_amdgcn_s_wait_tensorcnt(0);
    }
#else
    for (int c = threadIdx.x; c < DH_ * (KC_ / 8); c += 256) {
      int row = c / (KC_ / 8);
      int ch  = c % (KC_ / 8);
      v4u d = *(const v4u*)(const void*)(PTb + (size_t)row * N_ + kp + ch * 8);
      *(v4u*)(void*)(Bsm + row * LDB_ + ch * 8) = d;
    }
#endif
    __syncthreads();
#pragma unroll
    for (int kk = 0; kk < KC_; kk += 32) {
      const unsigned short* ar =
          Abh + (size_t)(m0 + l16) * 2048 + kp + kk + hp * 16;
      __builtin_prefetch(ar + 32, 0, 0);
      v16bf a = joinA(ar);
#pragma unroll
      for (int j = 0; j < 4; ++j) {
        v16bf b = loadB16(Bsm + (j * 16 + l16) * LDB_ + kk + hp * 16);
        acc[j] = wmma_bf16(a, b, acc[j]);
      }
    }
    __syncthreads();
  }

  unsigned short* Pr = PrOut + (size_t)bh * N_ * DH_;
  unsigned short* Pt = PTout + (size_t)bh * DH_ * N_;
#pragma unroll
  for (int j = 0; j < 4; ++j) {
    int col = j * 16 + l16;
    unsigned short t[8];
#pragma unroll
    for (int r = 0; r < 8; ++r) {
      unsigned short bfv = f2bf(acc[j][r]);
      t[r] = bfv;
      Pr[(size_t)(m0 + r + hp * 8) * DH_ + col] = bfv;   // row-major copy
    }
    v4u pk;
#pragma unroll
    for (int q = 0; q < 4; ++q)
      pk[q] = (unsigned)t[2 * q] | ((unsigned)t[2 * q + 1] << 16);
    *(v4u*)(void*)(Pt + (size_t)col * N_ + m0 + hp * 8) = pk;  // transposed copy
  }
}

// ---------------- H += (alpha * P @ W)  (1024 x 64 x 64) ----------------
// Wts holds alpha-scaled W^T in bf16. hop==0 initializes Hacc.
__global__ void filter_gemm(const unsigned short* __restrict__ Prow,
                            const unsigned short* __restrict__ Wts,
                            float* __restrict__ Hacc, int hop) {
  const int lane = threadIdx.x & 31, wv = threadIdx.x >> 5;
  const int l16 = lane & 15, hp = lane >> 4;
  const int bh = blockIdx.y;
  const int b = bh >> 3, h = bh & 7;
  const int m0 = (blockIdx.x * 8 + wv) * 16;
  const unsigned short* Pb = Prow + (size_t)bh * N_ * DH_;
  const unsigned short* Wt = Wts + (size_t)(h * 4 + hop) * DH_ * DH_;

  v8f zero = {};
  v8f acc[4] = {zero, zero, zero, zero};
  if (hop != 0) {
#pragma unroll
    for (int j = 0; j < 4; ++j) {
      int col = h * DH_ + j * 16 + l16;
#pragma unroll
      for (int r = 0; r < 8; ++r)
        acc[j][r] = Hacc[((size_t)b * N_ + (m0 + r + hp * 8)) * D_ + col];
    }
  }
#pragma unroll
  for (int k0 = 0; k0 < DH_; k0 += 32) {
    v16bf a = joinA(Pb + (size_t)(m0 + l16) * DH_ + k0 + hp * 16);
#pragma unroll
    for (int j = 0; j < 4; ++j) {
      v16bf bf = loadB(Wt + (size_t)(j * 16 + l16) * DH_ + k0 + hp * 16);
      acc[j] = wmma_bf16(a, bf, acc[j]);
    }
  }
#pragma unroll
  for (int j = 0; j < 4; ++j) {
    int col = h * DH_ + j * 16 + l16;
#pragma unroll
    for (int r = 0; r < 8; ++r)
      Hacc[((size_t)b * N_ + (m0 + r + hp * 8)) * D_ + col] = acc[j][r];
  }
}

// ---------------- out = Hcat @ Wproj^T + b  (4096 x 512 x 512) --------------
__global__ void proj_gemm(const unsigned short* __restrict__ Hb,
                          const unsigned short* __restrict__ Wp,
                          const float* __restrict__ bias,
                          float* __restrict__ out) {
  const int lane = threadIdx.x & 31, wv = threadIdx.x >> 5;
  const int l16 = lane & 15, hp = lane >> 4;
  const int tl = blockIdx.x * 8 + wv;              // [0,2048)
  const int m0 = (tl >> 3) * 16;
  const int n0 = (tl & 7) * 64;

  v8f zero = {};
  v8f acc[4] = {zero, zero, zero, zero};
  for (int k0 = 0; k0 < D_; k0 += 32) {
    const unsigned short* ar = Hb + (size_t)(m0 + l16) * D_ + k0 + hp * 16;
    __builtin_prefetch(ar + 32, 0, 0);
    v16bf a = joinA(ar);
#pragma unroll
    for (int j = 0; j < 4; ++j) {
      // B[c][o] = Wproj[o][c] -> column o is a contiguous row of Wp
      v16bf b = loadB(Wp + (size_t)(n0 + j * 16 + l16) * D_ + k0 + hp * 16);
      acc[j] = wmma_bf16(a, b, acc[j]);
    }
  }
#pragma unroll
  for (int j = 0; j < 4; ++j) {
    int n = n0 + j * 16 + l16;
    float bi = bias[n];
#pragma unroll
    for (int r = 0; r < 8; ++r)
      out[(size_t)(m0 + r + hp * 8) * D_ + n] = acc[j][r] + bi;
  }
}

// ---------------- host launcher ----------------
extern "C" void kernel_launch(void* const* d_in, const int* in_sizes, int n_in,
                              void* d_out, int out_size, void* d_ws, size_t ws_size,
                              hipStream_t stream) {
  const float* X      = (const float*)d_in[0];
  const float* temp   = (const float*)d_in[1];
  const float* Wf     = (const float*)d_in[2];
  const float* alpha  = (const float*)d_in[3];
  const float* Wproj  = (const float*)d_in[4];
  const float* bproj  = (const float*)d_in[5];
  const int*   pli    = (const int*)d_in[6];
  const int*   pLL    = (const int*)d_in[7];
  float* out = (float*)d_out;

  char* w = (char*)d_ws;
  size_t off = 0;
  auto take = [&](size_t bytes) -> char* {
    char* p = w + off;
    off = (off + bytes + 255) & ~(size_t)255;
    return p;
  };

  float*          S     = (float*)take((size_t)BH_ * N_ * N_ * 4);   // scores, A aliased bf16
  unsigned short* Xh    = (unsigned short*)take((size_t)BH_ * N_ * DH_ * 2);
  unsigned short* XhT   = (unsigned short*)take((size_t)BH_ * DH_ * N_ * 2);
  unsigned short* Prow1 = (unsigned short*)take((size_t)BH_ * N_ * DH_ * 2);
  unsigned short* PT1   = (unsigned short*)take((size_t)BH_ * DH_ * N_ * 2);
  unsigned short* Prow2 = (unsigned short*)take((size_t)BH_ * N_ * DH_ * 2);
  unsigned short* PT2   = (unsigned short*)take((size_t)BH_ * DH_ * N_ * 2);
  unsigned short* Wts   = (unsigned short*)take((size_t)H_ * 4 * DH_ * DH_ * 2);
  unsigned short* Wp    = (unsigned short*)take((size_t)D_ * D_ * 2);
  float*          Hacc  = (float*)take((size_t)B_ * N_ * D_ * 4);
  unsigned short* Hb    = (unsigned short*)take((size_t)B_ * N_ * D_ * 2);

  // 1) precision conversions / layout packs
  pack_x <<<(B_ * N_ * D_) / 256, 256, 0, stream>>>(X, Xh, XhT);
  pack_wf<<<(H_ * 4 * DH_ * DH_) / 256, 256, 0, stream>>>(Wf, alpha, Wts);
  pack_wp<<<(D_ * D_) / 256, 256, 0, stream>>>(Wproj, Wp);

  // 2) attention scores (bf16 WMMA), scaled by 1/(sqrt(DH)*tau)
  score_gemm<<<dim3(128, BH_), 256, 0, stream>>>(Xh, temp, S);

  // 3) exact top-k threshold + softmax; A stored bf16 in place
  softmax_topk<<<BH_ * N_, 256, 0, stream>>>(S, pli, pLL);

  // 4) hop-0 filter: H = alpha0 * Xh @ W0  (init)
  filter_gemm<<<dim3(8, BH_), 256, 0, stream>>>(Xh, Wts, Hacc, 0);

  // 5) hops 1..3: P <- A@P ; H += alpha_k * P @ Wk
  hop_gemm   <<<dim3(8, BH_), 256, 0, stream>>>(S, XhT, Prow1, PT1);
  filter_gemm<<<dim3(8, BH_), 256, 0, stream>>>(Prow1, Wts, Hacc, 1);
  hop_gemm   <<<dim3(8, BH_), 256, 0, stream>>>(S, PT1, Prow2, PT2);
  filter_gemm<<<dim3(8, BH_), 256, 0, stream>>>(Prow2, Wts, Hacc, 2);
  hop_gemm   <<<dim3(8, BH_), 256, 0, stream>>>(S, PT2, Prow1, PT1);
  filter_gemm<<<dim3(8, BH_), 256, 0, stream>>>(Prow1, Wts, Hacc, 3);

  // 6) projection
  cvt_h<<<(B_ * N_ * D_) / 256, 256, 0, stream>>>(Hacc, Hb);
  proj_gemm<<<256, 256, 0, stream>>>(Hb, Wp, bproj, out);
}
